// MambaLayer_57990648431347
// MI455X (gfx1250) — matrix operand order repaired and checked by reading, not compiled, not run on hardware
//
#include <hip/hip_runtime.h>
#include <hip/hip_bf16.h>
#include <math.h>

// ---------------- problem constants ----------------
#define NE    1024          // n_embd
#define DI    2048          // d_inner
#define DS    16            // d_state
#define DTR   64            // dt_rank
#define LSEQ  2048          // L
#define BB    2             // batch
#define TOK   (BB*LSEQ)     // 4096 tokens
#define XDBL  (DTR + 2*DS)  // 96

typedef __bf16 bf16_t;
typedef __attribute__((ext_vector_type(8)))  __bf16 v8bf;
typedef __attribute__((ext_vector_type(16))) __bf16 v16bf;
typedef __attribute__((ext_vector_type(8)))  float  v8f;

__device__ __forceinline__ unsigned short f32_to_bf16_bits(float f) {
    union { float f; unsigned int u; } v; v.f = f;
    unsigned int u = v.u;
    unsigned int r = u + 0x7FFFu + ((u >> 16) & 1u);   // round-to-nearest-even
    return (unsigned short)(r >> 16);
}

__device__ __forceinline__ float silu_f(float x) {
    return x / (1.f + __expf(-x));
}

// ---------------- elementwise f32 -> bf16 convert (weights) ----------------
__global__ void cvt_bf16_kernel(const float* __restrict__ src,
                                unsigned short* __restrict__ dst, int n) {
    int i = blockIdx.x * blockDim.x + threadIdx.x;
    if (i < n) dst[i] = f32_to_bf16_bits(src[i]);
}

// ---------------- LayerNorm -> bf16 activations ----------------
__global__ void ln_bf16_kernel(const float* __restrict__ x,
                               const float* __restrict__ g,
                               const float* __restrict__ b,
                               unsigned short* __restrict__ xn) {
    __shared__ float s0[256], s1[256];
    const int row = blockIdx.x;
    const int t = threadIdx.x;
    const float* xr = x + (size_t)row * NE;
    float sum = 0.f, sq = 0.f;
    for (int i = t; i < NE; i += 256) { float v = xr[i]; sum += v; sq += v * v; }
    s0[t] = sum; s1[t] = sq; __syncthreads();
    for (int off = 128; off > 0; off >>= 1) {
        if (t < off) { s0[t] += s0[t + off]; s1[t] += s1[t + off]; }
        __syncthreads();
    }
    const float mean = s0[0] * (1.f / NE);
    const float var  = s1[0] * (1.f / NE) - mean * mean;
    const float rstd = rsqrtf(var + 1e-5f);
    for (int i = t; i < NE; i += 256) {
        float v = (xr[i] - mean) * rstd * g[i] + b[i];
        xn[(size_t)row * NE + i] = f32_to_bf16_bits(v);
    }
}

// ---------------- WMMA bf16 GEMM:  C[M,N] = A[M,K] * W[N,K]^T ----------------
// Register-blocked: one wave computes a 32x64 C macro-tile (2 M-subtiles x
// 4 N-subtiles, 8 f32 accumulators).  Per 32-wide K-step: 2 A frags + 4 B
// frags feed 8 WMMAs (A reused 4x, B reused 2x).
// __launch_bounds__(128,1): allow the full VGPR budget so the 8 accumulators
// + operand frags + addresses fit without scratch spills in the hot loop.
// EPI: 0 = store, 1 = softplus(c + bias[n]), 2 = c + residual[m*ldc + n].
#define GEMM_WAVES 4
#define TILE_M 32
#define TILE_N 64
template <int EPI>
__global__ __launch_bounds__(32 * GEMM_WAVES, 1)
void gemm_wmma_bf16(const unsigned short* __restrict__ A,
                    const unsigned short* __restrict__ W,
                    float* __restrict__ C,
                    const float* __restrict__ aux,   // bias (EPI=1) / residual (EPI=2)
                    int M, int N, int K, int ldc) {
    const int lane  = threadIdx.x & 31;
    const int wave  = threadIdx.x >> 5;
    const int tileM = blockIdx.x * TILE_M;
    const int tileN = (blockIdx.y * GEMM_WAVES + wave) * TILE_N;
    if (tileN >= N) return;                       // wave-uniform: EXEC stays all-ones

    const int row = lane & 15;                    // M-row / N-row within 16-subtile
    const int sel = lane >> 4;                    // lane-group (K-half select)

    const unsigned short* a_ptr[2];
#pragma unroll
    for (int i = 0; i < 2; ++i)
        a_ptr[i] = A + (size_t)(tileM + i * 16 + row) * K;

    const unsigned short* w_ptr[4];
#pragma unroll
    for (int j = 0; j < 4; ++j) {
        int wr = tileN + j * 16 + row;
        if (wr > N - 1) wr = N - 1;               // clamp OOB reads (stores guarded below)
        w_ptr[j] = W + (size_t)wr * K;
    }

    v8f acc[2][4];
#pragma unroll
    for (int i = 0; i < 2; ++i)
#pragma unroll
        for (int j = 0; j < 4; ++j) acc[i][j] = (v8f){};

    for (int k0 = 0; k0 < K; k0 += 32) {
        v16bf a[2];
#pragma unroll
        for (int i = 0; i < 2; ++i) {
            // A 16x32 bf16 layout: lanes 0-15 hold K 0-7 & 16-23, lanes 16-31 hold 8-15 & 24-31
            v8bf alo = *(const v8bf*)(a_ptr[i] + k0 + sel * 8);
            v8bf ahi = *(const v8bf*)(a_ptr[i] + k0 + 16 + sel * 8);
            a[i] = __builtin_shufflevector(alo, ahi,
                       0,1,2,3,4,5,6,7,8,9,10,11,12,13,14,15);
        }
        v16bf bm[4];
#pragma unroll
        for (int j = 0; j < 4; ++j) {
            // B 32x16 bf16 layout: lanes 0-15 hold K 0-7 & 16-23, lanes 16-31 hold 8-15 & 24-31
            v8bf blo = *(const v8bf*)(w_ptr[j] + k0 + sel * 8);
            v8bf bhi = *(const v8bf*)(w_ptr[j] + k0 + 16 + sel * 8);
            bm[j] = __builtin_shufflevector(blo, bhi,
                       0,1,2,3,4,5,6,7,8,9,10,11,12,13,14,15);
        }
#pragma unroll
        for (int i = 0; i < 2; ++i)
#pragma unroll
            for (int j = 0; j < 4; ++j)
                acc[i][j] = __builtin_amdgcn_wmma_f32_16x16x32_bf16(
                                false, a[i], false, bm[j], (short)0, acc[i][j],
                                false, false);
    }

    // C/D layout: lane covers col N = lane&15; VGPR v covers row M = 8*(lane>>4)+v
#pragma unroll
    for (int j = 0; j < 4; ++j) {
        const int n = tileN + j * 16 + row;
        if (n >= N) continue;                     // uniform per 16-wide subtile
#pragma unroll
        for (int i = 0; i < 2; ++i) {
            const int m0 = tileM + i * 16 + sel * 8;
#pragma unroll
            for (int v = 0; v < 8; ++v) {
                const int m = m0 + v;
                float val = acc[i][j][v];
                if (EPI == 1) {
                    float z = val + aux[n];
                    val = (z > 20.f) ? z : log1pf(__expf(z));      // softplus
                } else if (EPI == 2) {
                    val += aux[(size_t)m * ldc + n];               // residual add
                }
                C[(size_t)m * ldc + n] = val;
            }
        }
    }
}

// ---------------- causal depthwise conv1d (taps=4) + SiLU ----------------
__global__ void conv_silu_kernel(const float* __restrict__ xz,   // [TOK, 2*DI], u = cols 0..DI-1
                                 const float* __restrict__ cw,   // [DI,1,4]
                                 const float* __restrict__ cb,   // [DI]
                                 float* __restrict__ uc,
                                 unsigned short* __restrict__ ucb) {
    const int idx = blockIdx.x * blockDim.x + threadIdx.x;       // TOK*DI
    if (idx >= TOK * DI) return;
    const int d = idx & (DI - 1);
    const int m = idx >> 11;          // /DI
    const int l = m & (LSEQ - 1);
    float acc = cb[d];
#pragma unroll
    for (int j = 0; j < 4; ++j) {
        const int ll = l - 3 + j;
        if (ll >= 0)
            acc += cw[d * 4 + j] * xz[(size_t)(m - 3 + j) * (2 * DI) + d];
    }
    const float sv = silu_f(acc);
    uc[idx]  = sv;
    ucb[idx] = f32_to_bf16_bits(sv);
}

// ---------------- extract dt columns of x_dbl as bf16 ----------------
__global__ void extract_dt_kernel(const float* __restrict__ xdbl,
                                  unsigned short* __restrict__ dtb) {
    const int idx = blockIdx.x * blockDim.x + threadIdx.x;       // TOK*DTR
    if (idx >= TOK * DTR) return;
    const int m = idx / DTR, r = idx - m * DTR;
    dtb[idx] = f32_to_bf16_bits(xdbl[(size_t)m * XDBL + r]);
}

// ---------------- selective scan: 16 lanes (state dims) per channel ----------------
// tid layout: [channel c = b*DI + d][s = lane16].  Each half-wave owns one
// channel; per step each lane does 1 exp + 2 fma, then a 4-level shfl_xor
// reduction produces y = sum_s h[s]*C[s].
__global__ void scan_kernel(const float* __restrict__ delta,   // [TOK, DI]
                            const float* __restrict__ uc,      // [TOK, DI]
                            const float* __restrict__ xdbl,    // [TOK, 96] (Bt @64, Ct @80)
                            const float* __restrict__ A_log,   // [DI, DS]
                            const float* __restrict__ Dp,      // [DI]
                            float* __restrict__ y) {           // [TOK, DI]
    const int tid = blockIdx.x * blockDim.x + threadIdx.x;     // BB*DI*DS threads
    if (tid >= BB * DI * DS) return;
    const int s = tid & (DS - 1);
    const int c = tid >> 4;            // channel = b*DI + d
    const int b = c / DI, d = c - b * DI;

    const float Ar = -__expf(A_log[d * DS + s]);
    const float Dd = Dp[d];
    float h = 0.f;

    for (int l = 0; l < LSEQ; ++l) {
        const size_t m  = (size_t)b * LSEQ + l;
        const float dv  = delta[m * DI + d];       // broadcast across 16 lanes
        const float uv  = uc[m * DI + d];
        const float Bv  = xdbl[m * XDBL + DTR + s];
        const float Cv  = xdbl[m * XDBL + DTR + DS + s];
        const float dA  = __expf(dv * Ar);
        h = dA * h + (dv * uv) * Bv;
        float p = h * Cv;
#pragma unroll
        for (int o = 8; o > 0; o >>= 1) p += __shfl_xor(p, o, DS);
        if (s == 0) y[m * DI + d] = p + uv * Dd;
    }
}

// ---------------- gate: yg = bf16( y * silu(res) ) ----------------
__global__ void gate_kernel(const float* __restrict__ y,
                            const float* __restrict__ xz,      // res = cols DI..2DI-1
                            unsigned short* __restrict__ yg) {
    const int idx = blockIdx.x * blockDim.x + threadIdx.x;     // TOK*DI
    if (idx >= TOK * DI) return;
    const int d = idx & (DI - 1);
    const int m = idx >> 11;
    const float r = xz[(size_t)m * (2 * DI) + DI + d];
    yg[idx] = f32_to_bf16_bits(y[idx] * silu_f(r));
}

// ---------------- launcher ----------------
static inline size_t align256(size_t v) { return (v + 255) & ~(size_t)255; }

extern "C" void kernel_launch(void* const* d_in, const int* in_sizes, int n_in,
                              void* d_out, int out_size, void* d_ws, size_t ws_size,
                              hipStream_t stream) {
    const float* x         = (const float*)d_in[0];
    const float* ln_g      = (const float*)d_in[1];
    const float* ln_b      = (const float*)d_in[2];
    const float* in_proj_w = (const float*)d_in[3];   // [2*DI, NE]
    const float* conv_w    = (const float*)d_in[4];   // [DI,1,4]
    const float* conv_b    = (const float*)d_in[5];
    const float* x_proj_w  = (const float*)d_in[6];   // [96, DI]
    const float* dt_proj_w = (const float*)d_in[7];   // [DI, DTR]
    const float* dt_proj_b = (const float*)d_in[8];
    const float* A_log     = (const float*)d_in[9];   // [DI, DS]
    const float* Dp        = (const float*)d_in[10];
    const float* out_proj_w= (const float*)d_in[11];  // [NE, DI]
    float* out = (float*)d_out;

    // workspace carve-up
    char* ws = (char*)d_ws; size_t off = 0;
    auto alloc = [&](size_t bytes) { char* p = ws + off; off = align256(off + bytes); return p; };
    unsigned short* xn_bf  = (unsigned short*)alloc((size_t)TOK * NE * 2);
    unsigned short* w_in   = (unsigned short*)alloc((size_t)2 * DI * NE * 2);
    unsigned short* w_xp   = (unsigned short*)alloc((size_t)XDBL * DI * 2);
    unsigned short* w_dt   = (unsigned short*)alloc((size_t)DI * DTR * 2);
    unsigned short* w_out  = (unsigned short*)alloc((size_t)NE * DI * 2);
    float*          xz     = (float*)alloc((size_t)TOK * 2 * DI * 4);
    float*          uc     = (float*)alloc((size_t)TOK * DI * 4);
    unsigned short* ucb    = (unsigned short*)alloc((size_t)TOK * DI * 2);
    float*          xdbl   = (float*)alloc((size_t)TOK * XDBL * 4);
    unsigned short* dtb    = (unsigned short*)alloc((size_t)TOK * DTR * 2);
    float*          delta  = (float*)alloc((size_t)TOK * DI * 4);
    float*          ybuf   = (float*)alloc((size_t)TOK * DI * 4);
    unsigned short* yg     = (unsigned short*)alloc((size_t)TOK * DI * 2);
    (void)ws_size; (void)n_in; (void)in_sizes; (void)out_size;

    const int CT = 256;
    auto cvt = [&](const float* s, unsigned short* d, int n) {
        cvt_bf16_kernel<<<(n + CT - 1) / CT, CT, 0, stream>>>(s, d, n);
    };
    // weight conversions (cheap, bandwidth-trivial)
    cvt(in_proj_w,  w_in,  2 * DI * NE);
    cvt(x_proj_w,   w_xp,  XDBL * DI);
    cvt(dt_proj_w,  w_dt,  DI * DTR);
    cvt(out_proj_w, w_out, NE * DI);

    // 1) LayerNorm -> bf16
    ln_bf16_kernel<<<TOK, 256, 0, stream>>>(x, ln_g, ln_b, xn_bf);

    const int NTB = TILE_N * GEMM_WAVES;   // N covered per block (256)

    // 2) in_proj:  xz[TOK, 2*DI] = xn * in_proj_w^T   (WMMA)
    {
        dim3 g(TOK / TILE_M, (2 * DI + NTB - 1) / NTB);
        gemm_wmma_bf16<0><<<g, 32 * GEMM_WAVES, 0, stream>>>(
            xn_bf, w_in, xz, nullptr, TOK, 2 * DI, NE, 2 * DI);
    }

    // 3) causal depthwise conv + SiLU
    conv_silu_kernel<<<(TOK * DI + CT - 1) / CT, CT, 0, stream>>>(xz, conv_w, conv_b, uc, ucb);

    // 4) x_proj:  x_dbl[TOK, 96] = u_c * x_proj_w^T   (WMMA)
    {
        dim3 g(TOK / TILE_M, (XDBL + NTB - 1) / NTB);
        gemm_wmma_bf16<0><<<g, 32 * GEMM_WAVES, 0, stream>>>(
            ucb, w_xp, xdbl, nullptr, TOK, XDBL, DI, XDBL);
    }

    // 5) dt_proj + softplus:  delta[TOK, DI]   (WMMA, fused epilogue)
    extract_dt_kernel<<<(TOK * DTR + CT - 1) / CT, CT, 0, stream>>>(xdbl, dtb);
    {
        dim3 g(TOK / TILE_M, (DI + NTB - 1) / NTB);
        gemm_wmma_bf16<1><<<g, 32 * GEMM_WAVES, 0, stream>>>(
            dtb, w_dt, delta, dt_proj_b, TOK, DI, DTR, DI);
    }

    // 6) selective scan: 16 lanes per channel, 65536 threads
    scan_kernel<<<(BB * DI * DS + CT - 1) / CT, CT, 0, stream>>>(
        delta, uc, xdbl, A_log, Dp, ybuf);

    // 7) gate, then out_proj with fused residual:  out = x + (y*silu(res)) * out_proj_w^T
    gate_kernel<<<(TOK * DI + CT - 1) / CT, CT, 0, stream>>>(ybuf, xz, yg);
    {
        dim3 g(TOK / TILE_M, (NE + NTB - 1) / NTB);
        gemm_wmma_bf16<2><<<g, 32 * GEMM_WAVES, 0, stream>>>(
            yg, w_out, out, x, TOK, NE, DI, NE);
    }
}
